// FPModule_326417514818
// MI455X (gfx1250) — matrix operand order repaired and checked by reading, not compiled, not run on hardware
//
#include <hip/hip_runtime.h>

typedef __attribute__((ext_vector_type(16))) _Float16 v16h;
typedef __attribute__((ext_vector_type(8)))  _Float16 v8h;
typedef __attribute__((ext_vector_type(4)))  _Float16 v4h;
typedef __attribute__((ext_vector_type(8)))  float    v8f;

#define B_    8
#define N_    2048
#define M_    8192
#define CIN   256
#define CSK   128
#define CH    384       // CIN + CSKIP
#define COUT  256
#define KSTEPS 12       // 384 / 32
#define H_STRIDE 392    // 384 + 8 halfs pad -> conflict-free ds_load_b128

// workspace layout (bytes)
#define WS_IDX_OFF 0u               // int  nidx[B*M*3]      = 786,432 B (in a 3 MiB slot)
#define WS_W_OFF   3145728u         // f32  nw  [B*M*3]      = 786,432 B (in a 3 MiB slot)
#define WS_WH_OFF  6291456u         // f16  whFrag[12*16*32*16] = 196,608 B

static __device__ __forceinline__ unsigned umin_(unsigned a, unsigned b) { return a < b ? a : b; }
static __device__ __forceinline__ unsigned umax_(unsigned a, unsigned b) { return a > b ? a : b; }

// ---------------------------------------------------------------------------
// Kernel 0: convert W (f32 [384 x 256] row-major) into f16 WMMA B-fragments.
// Fragment order: [kstep t][col-tile ct][lane] x 16 halfs (contiguous 32 B).
// B 32x16 order chosen here (contiguous, matches A): lane = n%16 + 16*half, element j -> K = half*16 + j.
// ---------------------------------------------------------------------------
__global__ __launch_bounds__(256) void pack_w_kernel(const float* __restrict__ W,
                                                     _Float16* __restrict__ whFrag) {
    int tid  = blockIdx.x * 256 + threadIdx.x;      // 0 .. 6143
    int t    = tid >> 9;                            // k-step (12)
    int rem  = tid & 511;
    int ct   = rem >> 5;                            // col tile (16)
    int lane = rem & 31;
    int n    = ct * 16 + (lane & 15);
    int half = lane >> 4;
    v16h v;
#pragma unroll
    for (int j = 0; j < 16; ++j) {
        int k = t * 32 + half * 16 + j;
        v[j] = (_Float16)W[k * COUT + n];
    }
    *(v16h*)(whFrag + (size_t)tid * 16) = v;
}

// ---------------------------------------------------------------------------
// Kernel 1: per-cloud 3-NN, branchless packed-key top-3.
// key = (bits(d2) & ~0x7FF) | src_index  (d2 >= 0 so u32 order == float order)
// 5 v_min/v_max_u32 per point maintain the sorted top-3; exact distances for
// the winners are recomputed from LDS positions afterwards (matches reference
// d2e exactly). Source positions interleaved xyz -> 3 ds_load_b128 / 4 points.
// ---------------------------------------------------------------------------
__global__ __launch_bounds__(256) void knn_kernel(const float* __restrict__ pos,
                                                  const float* __restrict__ pos_skip,
                                                  int*   __restrict__ nidx,
                                                  float* __restrict__ nw) {
    __shared__ __attribute__((aligned(16))) float spt[N_ * 3];   // 24 KB
    int b = blockIdx.x >> 5;                        // 32 blocks per cloud
    int m = ((blockIdx.x & 31) << 8) + threadIdx.x; // target in cloud

    const float4* ps4 = (const float4*)(pos + (size_t)b * N_ * 3);
    for (int i = threadIdx.x; i < (N_ * 3) / 4; i += 256)
        ((float4*)spt)[i] = ps4[i];
    __syncthreads();

    const float* pq = pos_skip + ((size_t)b * M_ + m) * 3;
    float qx = pq[0], qy = pq[1], qz = pq[2];

    unsigned k0 = 0xFFFFFFFFu, k1 = 0xFFFFFFFFu, k2 = 0xFFFFFFFFu;
    for (int n = 0; n < N_; n += 4) {
        float4 p0 = *(const float4*)&spt[n * 3 + 0];
        float4 p1 = *(const float4*)&spt[n * 3 + 4];
        float4 p2 = *(const float4*)&spt[n * 3 + 8];
        float px[4] = { p0.x, p0.w, p1.z, p2.y };
        float py[4] = { p0.y, p1.x, p1.w, p2.z };
        float pz[4] = { p0.z, p1.y, p2.x, p2.w };
#pragma unroll
        for (int u = 0; u < 4; ++u) {
            float dx = qx - px[u], dy = qy - py[u], dz = qz - pz[u];
            float d = fmaf(dx, dx, fmaf(dy, dy, dz * dz));
            unsigned key = (__float_as_uint(d) & 0xFFFFF800u) | (unsigned)(n + u);
            unsigned lo = umin_(key, k0);
            unsigned hi = umax_(key, k0);
            k0 = lo;
            unsigned lo1 = umin_(hi, k1);
            unsigned hi1 = umax_(hi, k1);
            k1 = lo1;
            k2 = umin_(hi1, k2);
        }
    }
    int i0 = (int)(k0 & 2047u), i1 = (int)(k1 & 2047u), i2 = (int)(k2 & 2047u);

    // exact inverse squared-distance weights (reference's d2e), normalized
    float e0, e1, e2;
    {
        float dx = qx - spt[i0 * 3], dy = qy - spt[i0 * 3 + 1], dz = qz - spt[i0 * 3 + 2];
        e0 = fmaf(dx, dx, fmaf(dy, dy, dz * dz));
        dx = qx - spt[i1 * 3]; dy = qy - spt[i1 * 3 + 1]; dz = qz - spt[i1 * 3 + 2];
        e1 = fmaf(dx, dx, fmaf(dy, dy, dz * dz));
        dx = qx - spt[i2 * 3]; dy = qy - spt[i2 * 3 + 1]; dz = qz - spt[i2 * 3 + 2];
        e2 = fmaf(dx, dx, fmaf(dy, dy, dz * dz));
    }
    float w0 = 1.0f / fmaxf(e0, 1e-16f);
    float w1 = 1.0f / fmaxf(e1, 1e-16f);
    float w2 = 1.0f / fmaxf(e2, 1e-16f);
    float inv = 1.0f / (w0 + w1 + w2);
    size_t r = (size_t)b * M_ + m;
    nidx[r * 3 + 0] = b * N_ + i0;
    nidx[r * 3 + 1] = b * N_ + i1;
    nidx[r * 3 + 2] = b * N_ + i2;
    nw[r * 3 + 0] = w0 * inv;
    nw[r * 3 + 1] = w1 * inv;
    nw[r * 3 + 2] = w2 * inv;
}

// ---------------------------------------------------------------------------
// Kernel 2: fused interpolate + concat + GEMM (WMMA f16->f32) + bias + leaky.
// Block: 256 threads (8 waves), computes 64 rows x 256 cols of the output.
// A 16x32 f16 order chosen here (contiguous, matches B): lane = row%16 + 16*half; element j -> K = half*16 + j
//   => two contiguous 16 B runs per lane -> ds_load_b128 x2.
// B fragments ping-pong through two register sets so load(j+1) overlaps wmma(j).
// ---------------------------------------------------------------------------
__global__ __launch_bounds__(256) void fp_fused_kernel(const float* __restrict__ x,
                                                       const float* __restrict__ x_skip,
                                                       const float* __restrict__ bias,
                                                       const int*   __restrict__ nidx,
                                                       const float* __restrict__ nw,
                                                       const _Float16* __restrict__ whFrag,
                                                       float* __restrict__ out) {
    __shared__ __attribute__((aligned(16))) _Float16 hLds[64 * H_STRIDE]; // ~50 KB
    const int tid = threadIdx.x;
    const int rowBase = blockIdx.x * 64;            // global output row base

    // ---- Phase A: gather + interpolate xi (f16) into LDS ----
    const float4* X4 = (const float4*)x;            // [B*N][64] float4
    for (int e = tid; e < 64 * 64; e += 256) {      // 64 rows x 64 float4-chunks
        int r  = e >> 6;
        int c4 = e & 63;
        size_t row = (size_t)(rowBase + r);
        int  j0 = nidx[row * 3 + 0], j1 = nidx[row * 3 + 1], j2 = nidx[row * 3 + 2];
        float w0 = nw[row * 3 + 0], w1 = nw[row * 3 + 1], w2 = nw[row * 3 + 2];
        float4 a = X4[(size_t)j0 * 64 + c4];
        float4 bq= X4[(size_t)j1 * 64 + c4];
        float4 c = X4[(size_t)j2 * 64 + c4];
        v4h h;
        h[0] = (_Float16)(w0 * a.x + w1 * bq.x + w2 * c.x);
        h[1] = (_Float16)(w0 * a.y + w1 * bq.y + w2 * c.y);
        h[2] = (_Float16)(w0 * a.z + w1 * bq.z + w2 * c.z);
        h[3] = (_Float16)(w0 * a.w + w1 * bq.w + w2 * c.w);
        *(v4h*)&hLds[r * H_STRIDE + c4 * 4] = h;    // 8 B aligned
    }
    // skip features (f16) into LDS columns [256, 384)
    const float4* XS4 = (const float4*)x_skip;      // [B*M][32] float4
    for (int e = tid; e < 64 * 32; e += 256) {
        int r  = e >> 5;
        int c4 = e & 31;
        float4 a = XS4[(size_t)(rowBase + r) * 32 + c4];
        v4h h;
        h[0] = (_Float16)a.x; h[1] = (_Float16)a.y;
        h[2] = (_Float16)a.z; h[3] = (_Float16)a.w;
        *(v4h*)&hLds[r * H_STRIDE + CIN + c4 * 4] = h;
    }
    __syncthreads();

    // ---- Phase B: WMMA main loop ----
    const int lane    = tid & 31;
    const int wv      = tid >> 5;
    const int strip   = wv >> 1;                    // row strip 0..3  (16 rows)
    const int colHalf = wv & 1;                     // col tiles colHalf*8 .. +8
    const int rowL    = strip * 16 + (lane & 15);
    const int half    = lane >> 4;

    v8f acc[8] = {};

    for (int t = 0; t < KSTEPS; ++t) {
        // A fragment: two 16 B contiguous LDS loads per lane
        const int kb = t * 32 + half * 16;
        v8h alo = *(const v8h*)&hLds[rowL * H_STRIDE + kb];
        v8h ahi = *(const v8h*)&hLds[rowL * H_STRIDE + kb + 8];
        v16h afrag;
#pragma unroll
        for (int i = 0; i < 8; ++i) { afrag[i] = alo[i]; afrag[i + 8] = ahi[i]; }

        const _Float16* wbase =
            whFrag + ((size_t)(t * 16 + colHalf * 8) * 32 + lane) * 16;
        v16h bcur = *(const v16h*)wbase;            // tile j = 0
#pragma unroll
        for (int j = 0; j < 8; ++j) {
            v16h bnxt = bcur;
            if (j < 7) bnxt = *(const v16h*)(wbase + (size_t)(j + 1) * 32 * 16);
            acc[j] = __builtin_amdgcn_wmma_f32_16x16x32_f16(
                false, afrag, false, bcur, (short)0, acc[j], false, false);
            bcur = bnxt;
        }
    }

    // ---- Epilogue: bias + leaky relu, f32 store ----
#pragma unroll
    for (int j = 0; j < 8; ++j) {
        int ct  = colHalf * 8 + j;
        int col = ct * 16 + (lane & 15);
        float bb = bias[col];
#pragma unroll
        for (int v = 0; v < 8; ++v) {
            int rl = strip * 16 + v + half * 8;     // C/D layout: VGPR v -> M=v / M=v+8
            float val = acc[j][v] + bb;
            val = fmaxf(val, 0.01f * val);          // leaky_relu(0.01)
            out[(size_t)(rowBase + rl) * COUT + col] = val;
        }
    }
}

// ---------------------------------------------------------------------------
// Kernel 3: tuple tail — pos_skip passthrough, batch_skip as float.
// ---------------------------------------------------------------------------
__global__ __launch_bounds__(256) void tail_kernel(const float* __restrict__ pos_skip,
                                                   const int*   __restrict__ batch_skip,
                                                   float* __restrict__ out_pos,
                                                   float* __restrict__ out_batch) {
    int i = blockIdx.x * 256 + threadIdx.x;
    if (i < B_ * M_ * 3) out_pos[i] = pos_skip[i];
    if (i < B_ * M_)     out_batch[i] = (float)batch_skip[i];
}

extern "C" void kernel_launch(void* const* d_in, const int* in_sizes, int n_in,
                              void* d_out, int out_size, void* d_ws, size_t ws_size,
                              hipStream_t stream) {
    const float* x         = (const float*)d_in[0];
    const float* pos       = (const float*)d_in[1];
    /* d_in[2] = batch (unused; implicit in equal-size reshape) */
    const float* x_skip    = (const float*)d_in[3];
    const float* pos_skip  = (const float*)d_in[4];
    const int*   batch_sk  = (const int*)d_in[5];
    const float* W         = (const float*)d_in[6];
    const float* bias      = (const float*)d_in[7];

    char* ws = (char*)d_ws;
    int*      nidx   = (int*)(ws + WS_IDX_OFF);
    float*    nw     = (float*)(ws + WS_W_OFF);
    _Float16* whFrag = (_Float16*)(ws + WS_WH_OFF);

    float* out       = (float*)d_out;
    float* out_pos   = out + (size_t)B_ * M_ * COUT;
    float* out_batch = out_pos + (size_t)B_ * M_ * 3;

    pack_w_kernel<<<24, 256, 0, stream>>>(W, whFrag);
    knn_kernel<<<B_ * (M_ / 256), 256, 0, stream>>>(pos, pos_skip, nidx, nw);
    fp_fused_kernel<<<(B_ * M_) / 64, 256, 0, stream>>>(x, x_skip, bias, nidx, nw,
                                                        whFrag, out);
    tail_kernel<<<(B_ * M_ * 3 + 255) / 256, 256, 0, stream>>>(pos_skip, batch_sk,
                                                               out_pos, out_batch);
}
